// MultiHeadAttention_51573967290449
// MI455X (gfx1250) — matrix operand order repaired and checked by reading, not compiled, not run on hardware
//
#include <hip/hip_runtime.h>

// ---------------------------------------------------------------------------
// MHA forward for MI455X (gfx1250): bf16 WMMA everywhere, fp32 accumulate.
// B=4, S=2048, D_MODEL=1024, H=16, D_K=64.
// Double-buffered LDS tiles with async global->LDS copies (ASYNCcnt) when the
// toolchain exposes them; synchronous fallback otherwise.
// ---------------------------------------------------------------------------

typedef __bf16 bf16_t;
typedef __attribute__((ext_vector_type(16))) __bf16 v16bf;
typedef __attribute__((ext_vector_type(8)))  __bf16 v8bf;
typedef __attribute__((ext_vector_type(8)))  float  v8f;

#define BDIM  4
#define SDIM  2048
#define DMODEL 1024
#define HEADS 16
#define DK    64
#define MTOT  (BDIM * SDIM)     // 8192 rows

// ----- async global->LDS copy (CDNA5), with sync fallback -------------------
#if __has_builtin(__builtin_amdgcn_global_load_async_to_lds_b128)
#define USE_ASYNC_LDS 1
#endif

typedef int async_v4i __attribute__((vector_size(16)));
typedef __attribute__((address_space(1))) async_v4i* gptr_v4i;
typedef __attribute__((address_space(3))) async_v4i* lptr_v4i;

__device__ __forceinline__ void cp_b128(void* dst_lds, const void* src_gbl) {
#ifdef USE_ASYNC_LDS
  __builtin_amdgcn_global_load_async_to_lds_b128(
      (gptr_v4i)(src_gbl), (lptr_v4i)(dst_lds), 0, 0);
#else
  *(uint4*)dst_lds = *(const uint4*)src_gbl;
#endif
}

#ifdef USE_ASYNC_LDS
#if __has_builtin(__builtin_amdgcn_s_wait_asynccnt)
#define WAIT_ASYNC(n) __builtin_amdgcn_s_wait_asynccnt(n)
#else
#define WAIT_ASYNC(n) asm volatile("s_wait_asynccnt %0" ::"i"(n))
#endif
#else
#define WAIT_ASYNC(n) ((void)0)
#endif

// ----- fp32 -> bf16 (round to nearest even) --------------------------------
__device__ __forceinline__ unsigned short bf16_bits(float f) {
  unsigned u = __builtin_bit_cast(unsigned, f);
  unsigned r = u + 0x7FFFu + ((u >> 16) & 1u);
  return (unsigned short)(r >> 16);
}
__device__ __forceinline__ bf16_t to_bf16(float f) {
  unsigned short s = bf16_bits(f);
  return __builtin_bit_cast(bf16_t, s);
}

// Build a 16-element bf16 WMMA fragment from two 16B-aligned 8-element chunks.
__device__ __forceinline__ v16bf ld_frag(const bf16_t* p0, const bf16_t* p1) {
  v8bf a = *(const v8bf*)p0;
  v8bf b = *(const v8bf*)p1;
  v16bf r;
#pragma unroll
  for (int i = 0; i < 8; ++i) { r[i] = a[i]; r[i + 8] = b[i]; }
  return r;
}

// ----- cast kernel: fp32 -> bf16, 4 elements / thread ----------------------
__global__ void cast_f32_bf16_k(const float4* __restrict__ src,
                                ushort4* __restrict__ dst, int n4) {
  int i = blockIdx.x * blockDim.x + threadIdx.x;
  if (i < n4) {
    float4 f = src[i];
    ushort4 o;
    o.x = bf16_bits(f.x); o.y = bf16_bits(f.y);
    o.z = bf16_bits(f.z); o.w = bf16_bits(f.w);
    dst[i] = o;
  }
}

// ----- tiled bf16 WMMA GEMM: C[m][n] = sum_k A[m][k]*Bm[n][k] + bias[n] ----
// 256 threads = 8 waves; block tile 128x128; per-wave tile 32x64 (8 WMMA/step)
// Store modes: 1 = bf16 [b,h,s,d] head layout (Q,K)
//              2 = bf16 [b,h,d,s] transposed   (V)
//              3 = fp32 [m][n]                 (final output)
#define LDA 40   // padded LDS stride (bf16 elems) for 32-wide K tiles
#define GEMM_NK (DMODEL / 32)
__global__ __launch_bounds__(256)
void gemm_bf16_wmma(const bf16_t* __restrict__ A,
                    const bf16_t* __restrict__ Bm,
                    const float*  __restrict__ bias,
                    void* __restrict__ out, int mode) {
  __shared__ bf16_t sA[2][128 * LDA];
  __shared__ bf16_t sB[2][128 * LDA];

  const int tile_n = blockIdx.x;          // N/128 = 8
  const int tile_m = blockIdx.y;          // M/128 = 64
  const int tid    = threadIdx.x;
  const int wave   = tid >> 5;
  const int lane   = tid & 31;
  const int l16    = lane & 15;
  const int lhalf  = lane >> 4;
  const int wave_m = wave & 3;            // 32-row strip
  const int wave_n = wave >> 2;           // 64-col strip

  const int row_a0 = tile_m * 128;
  const int col_b0 = tile_n * 128;

  const int ldr = tid >> 1;               // 0..127 : row staged by this thread
  const int ldh = (tid & 1) * 16;         // k-half within 32-wide tile

  v8f acc[2][4] = {};

  // stage one 128x32 bf16 tile pair into LDS buffer `buf`
  auto stage = [&](int buf, int k0) {
#pragma unroll
    for (int j = 0; j < 2; ++j) {
      cp_b128(sA[buf] + ldr * LDA + ldh + j * 8,
              A  + (size_t)(row_a0 + ldr) * DMODEL + k0 + ldh + j * 8);
      cp_b128(sB[buf] + ldr * LDA + ldh + j * 8,
              Bm + (size_t)(col_b0 + ldr) * DMODEL + k0 + ldh + j * 8);
    }
  };

  stage(0, 0);                            // prologue

  for (int ks = 0; ks < GEMM_NK; ++ks) {
    const int cur = ks & 1;
    if (ks + 1 < GEMM_NK) {               // prefetch next tile into other buf
      stage(1 - cur, (ks + 1) * 32);
      WAIT_ASYNC(4);                      // current tile landed; next in flight
    } else {
      WAIT_ASYNC(0);
    }
    __syncthreads();

    v16bf afrag[2];
#pragma unroll
    for (int fa = 0; fa < 2; ++fa) {
      const bf16_t* ap =
          sA[cur] + (wave_m * 32 + fa * 16 + l16) * LDA + lhalf * 16;
      afrag[fa] = ld_frag(ap, ap + 8);
    }
#pragma unroll
    for (int nt = 0; nt < 4; ++nt) {
      const bf16_t* bp =
          sB[cur] + (wave_n * 64 + nt * 16 + l16) * LDA + lhalf * 16;
      v16bf bfrag = ld_frag(bp, bp + 8);
#pragma unroll
      for (int fa = 0; fa < 2; ++fa)
        acc[fa][nt] = __builtin_amdgcn_wmma_f32_16x16x32_bf16(
            false, afrag[fa], false, bfrag, (short)0, acc[fa][nt], false, false);
    }
    __syncthreads();
  }

  // epilogue: bias + layout-dependent store
  float bb[4];
#pragma unroll
  for (int nt = 0; nt < 4; ++nt)
    bb[nt] = bias[col_b0 + wave_n * 64 + nt * 16 + l16];

#pragma unroll
  for (int fa = 0; fa < 2; ++fa) {
#pragma unroll
    for (int nt = 0; nt < 4; ++nt) {
#pragma unroll
      for (int r = 0; r < 8; ++r) {
        float v = acc[fa][nt][r] + bb[nt];
        int m = row_a0 + wave_m * 32 + fa * 16 + lhalf * 8 + r;
        int n = col_b0 + wave_n * 64 + nt * 16 + l16;
        if (mode == 3) {
          ((float*)out)[(size_t)m * DMODEL + n] = v;
        } else {
          int bI = m >> 11, sI = m & (SDIM - 1);
          int hI = n >> 6,  dI = n & (DK - 1);
          size_t off = (mode == 1)
            ? (((size_t)(bI * HEADS + hI) * SDIM) + sI) * DK + dI
            : (((size_t)(bI * HEADS + hI) * DK)   + dI) * SDIM + sI;
          ((bf16_t*)out)[off] = to_bf16(v);
        }
      }
    }
  }
}

// ----- flash-attention: one block per (b, h, 64-row query tile) ------------
#define LDT 72   // padded LDS stride (bf16 elems) for 64-wide tiles
#define NKT (SDIM / 64)
__global__ __launch_bounds__(128)
void attn_fwd_wmma(const bf16_t* __restrict__ Qh,   // [B*H][S][64]
                   const bf16_t* __restrict__ Kh,   // [B*H][S][64]
                   const bf16_t* __restrict__ Vt,   // [B*H][64][S]
                   const int*    __restrict__ mask, // [B][S]
                   bf16_t* __restrict__ O) {        // [B][S][1024]
  __shared__ bf16_t sQ[64 * LDT];
  __shared__ bf16_t sK[2][64 * LDT];
  __shared__ bf16_t sV[2][64 * LDT];
  __shared__ float  sS[4][16][66];
  __shared__ bf16_t sP[4][16 * LDT];
  __shared__ float  sM[64], sL[64], sFac[64];

  const int qt = blockIdx.x, h = blockIdx.y, b = blockIdx.z;
  const size_t bh = (size_t)(b * HEADS + h);
  const bf16_t* Qb = Qh + bh * (SDIM * DK);
  const bf16_t* Kb = Kh + bh * (SDIM * DK);
  const bf16_t* Vb = Vt + bh * (DK * SDIM);

  const int tid   = threadIdx.x;
  const int wave  = tid >> 5;
  const int lane  = tid & 31;
  const int l16   = lane & 15;
  const int lhalf = lane >> 4;
  const int q0    = qt * 64;

  const int ldr = tid >> 1;               // 0..63
  const int ldh = (tid & 1) * 32;         // 32-element half of 64-wide row

  // async-stage one 64x64 K tile + 64x64 V tile into buffer `buf`
  auto stageKV = [&](int buf, int k0) {
#pragma unroll
    for (int j = 0; j < 4; ++j) {
      cp_b128(sK[buf] + ldr * LDT + ldh + j * 8,
              Kb + (size_t)(k0 + ldr) * DK + ldh + j * 8);
      cp_b128(sV[buf] + ldr * LDT + ldh + j * 8,
              Vb + (size_t)ldr * SDIM + k0 + ldh + j * 8);
    }
  };

  // stage Q tile (64x64 bf16, reused every iteration), init running stats
  {
    const uint4* g = (const uint4*)(Qb + (size_t)(q0 + ldr) * DK + ldh);
    uint4* d = (uint4*)(sQ + ldr * LDT + ldh);
    d[0] = g[0]; d[1] = g[1]; d[2] = g[2]; d[3] = g[3];
  }
  if (tid < 64) { sM[tid] = -3.0e38f; sL[tid] = 0.0f; }

  stageKV(0, 0);                          // prologue

  v8f accO[4] = {};
  const float scale = 0.125f;             // 1/sqrt(64)

  for (int kt = 0; kt < NKT; ++kt) {
    const int k0  = kt * 64;
    const int cur = kt & 1;
    if (kt + 1 < NKT) {                   // prefetch next K/V tile
      stageKV(1 - cur, (kt + 1) * 64);
      WAIT_ASYNC(8);
    } else {
      WAIT_ASYNC(0);
    }
    __syncthreads();

    // scores: wave's 16 q-rows x 64 keys  (K-dim = DK = 64 -> 2 wmma steps)
    v8f sc[4] = {};
#pragma unroll
    for (int kk = 0; kk < DK; kk += 32) {
      const bf16_t* ap = sQ + (wave * 16 + l16) * LDT + kk + lhalf * 16;
      v16bf aq = ld_frag(ap, ap + 8);
#pragma unroll
      for (int nt = 0; nt < 4; ++nt) {
        const bf16_t* bp = sK[cur] + (nt * 16 + l16) * LDT + kk + lhalf * 16;
        v16bf bk = ld_frag(bp, bp + 8);
        sc[nt] = __builtin_amdgcn_wmma_f32_16x16x32_bf16(
            false, aq, false, bk, (short)0, sc[nt], false, false);
      }
    }
    // spill score tile to LDS (row = vgpr + 8*lhalf, col = nt*16 + l16)
#pragma unroll
    for (int nt = 0; nt < 4; ++nt)
#pragma unroll
      for (int r = 0; r < 8; ++r)
        sS[wave][lhalf * 8 + r][nt * 16 + l16] = sc[nt][r];
    __syncthreads();

    // online softmax: 2 lanes per row, 32 cols each
    {
      const int lr = lane >> 1;
      const int c0 = (lane & 1) * 32;
      float mloc = -3.0e38f;
      float xs[32];
#pragma unroll
      for (int c = 0; c < 32; ++c) {
        float x = sS[wave][lr][c0 + c] * scale;
        if (mask[b * SDIM + k0 + c0 + c] == 0) x = -1.0e9f;
        xs[c] = x;
        mloc = fmaxf(mloc, x);
      }
      mloc = fmaxf(mloc, __shfl_xor(mloc, 1, 32));
      const int grow = wave * 16 + lr;
      float mold = sM[grow];
      float mnew = fmaxf(mold, mloc);
      float fac  = __expf(mold - mnew);
      float psum = 0.0f;
#pragma unroll
      for (int c = 0; c < 32; ++c) {
        float p = __expf(xs[c] - mnew);
        psum += p;
        sP[wave][lr * LDT + c0 + c] = to_bf16(p);
      }
      psum += __shfl_xor(psum, 1, 32);
      if ((lane & 1) == 0) {
        sM[grow]   = mnew;
        sL[grow]   = sL[grow] * fac + psum;
        sFac[grow] = fac;
      }
    }
    __syncthreads();

    // rescale running output, then accumulate P x V
    {
      float fac[8];
#pragma unroll
      for (int r = 0; r < 8; ++r) fac[r] = sFac[wave * 16 + lhalf * 8 + r];
#pragma unroll
      for (int nt = 0; nt < 4; ++nt)
#pragma unroll
        for (int r = 0; r < 8; ++r) accO[nt][r] *= fac[r];
    }
#pragma unroll
    for (int kk = 0; kk < 64; kk += 32) {
      const bf16_t* ap = sP[wave] + l16 * LDT + kk + lhalf * 16;
      v16bf pf = ld_frag(ap, ap + 8);
#pragma unroll
      for (int nt = 0; nt < 4; ++nt) {
        const bf16_t* bp = sV[cur] + (nt * 16 + l16) * LDT + kk + lhalf * 16;
        v16bf vf = ld_frag(bp, bp + 8);
        accO[nt] = __builtin_amdgcn_wmma_f32_16x16x32_bf16(
            false, pf, false, vf, (short)0, accO[nt], false, false);
      }
    }
    __syncthreads();                      // all reads of buf `cur` done
  }

  // normalize and store: O[b][s][h*64 + d], bf16
  float inv[8];
#pragma unroll
  for (int r = 0; r < 8; ++r) inv[r] = 1.0f / sL[wave * 16 + lhalf * 8 + r];
#pragma unroll
  for (int nt = 0; nt < 4; ++nt)
#pragma unroll
    for (int r = 0; r < 8; ++r) {
      int srow = q0 + wave * 16 + lhalf * 8 + r;
      int e = h * DK + nt * 16 + l16;
      O[((size_t)b * SDIM + srow) * DMODEL + e] = to_bf16(accO[nt][r] * inv[r]);
    }
}

// ---------------------------------------------------------------------------
extern "C" void kernel_launch(void* const* d_in, const int* in_sizes, int n_in,
                              void* d_out, int out_size, void* d_ws, size_t ws_size,
                              hipStream_t stream) {
  const float* x    = (const float*)d_in[0];
  const int*   mask = (const int*)  d_in[1];
  const float* Wq   = (const float*)d_in[2];
  const float* bq   = (const float*)d_in[3];
  const float* Wk   = (const float*)d_in[4];
  const float* bk   = (const float*)d_in[5];
  const float* Wv   = (const float*)d_in[6];
  const float* bv   = (const float*)d_in[7];
  const float* Wo   = (const float*)d_in[8];
  const float* bo   = (const float*)d_in[9];

  char* ws = (char*)d_ws;
  bf16_t* xb  = (bf16_t*)ws;  ws += (size_t)MTOT * DMODEL * 2;          // 16 MB
  bf16_t* wqb = (bf16_t*)ws;  ws += (size_t)DMODEL * DMODEL * 2;        //  2 MB
  bf16_t* wkb = (bf16_t*)ws;  ws += (size_t)DMODEL * DMODEL * 2;
  bf16_t* wvb = (bf16_t*)ws;  ws += (size_t)DMODEL * DMODEL * 2;
  bf16_t* wob = (bf16_t*)ws;  ws += (size_t)DMODEL * DMODEL * 2;
  bf16_t* Qh  = (bf16_t*)ws;  ws += (size_t)MTOT * DMODEL * 2;          // 16 MB
  bf16_t* Kh  = (bf16_t*)ws;  ws += (size_t)MTOT * DMODEL * 2;
  bf16_t* Vt  = (bf16_t*)ws;  ws += (size_t)MTOT * DMODEL * 2;
  bf16_t* Ob  = (bf16_t*)ws;  ws += (size_t)MTOT * DMODEL * 2;

  // 1) casts to bf16
  {
    int nx4 = MTOT * DMODEL / 4;
    cast_f32_bf16_k<<<nx4 / 256, 256, 0, stream>>>((const float4*)x, (ushort4*)xb, nx4);
    int nw4 = DMODEL * DMODEL / 4;
    cast_f32_bf16_k<<<nw4 / 256, 256, 0, stream>>>((const float4*)Wq, (ushort4*)wqb, nw4);
    cast_f32_bf16_k<<<nw4 / 256, 256, 0, stream>>>((const float4*)Wk, (ushort4*)wkb, nw4);
    cast_f32_bf16_k<<<nw4 / 256, 256, 0, stream>>>((const float4*)Wv, (ushort4*)wvb, nw4);
    cast_f32_bf16_k<<<nw4 / 256, 256, 0, stream>>>((const float4*)Wo, (ushort4*)wob, nw4);
  }

  // 2) Q/K/V projections (WMMA GEMM), head-major stores; V transposed
  dim3 gg(DMODEL / 128, MTOT / 128);
  gemm_bf16_wmma<<<gg, 256, 0, stream>>>(xb, wqb, bq, Qh, 1);
  gemm_bf16_wmma<<<gg, 256, 0, stream>>>(xb, wkb, bk, Kh, 1);
  gemm_bf16_wmma<<<gg, 256, 0, stream>>>(xb, wvb, bv, Vt, 2);

  // 3) flash attention
  dim3 ga(SDIM / 64, HEADS, BDIM);
  attn_fwd_wmma<<<ga, 128, 0, stream>>>(Qh, Kh, Vt, mask, Ob);

  // 4) output projection -> fp32 d_out
  gemm_bf16_wmma<<<gg, 256, 0, stream>>>(Ob, wob, bo, d_out, 3);
}
